// MultiHeadAttention_22900765622848
// MI455X (gfx1250) — matrix operand order repaired and checked by reading, not compiled, not run on hardware
//
#include <hip/hip_runtime.h>
#include <stdint.h>

typedef __attribute__((ext_vector_type(16))) _Float16 v16h;
typedef __attribute__((ext_vector_type(8)))  float    v8f;

#define D_MODEL_ 1024
#define N_HEADS_ 16
#define D_K_     64
#define BATCH_   4
#define SEQ_     2048

union Frag16 { v16h v; uint32_t u[8]; };
union Cv4    { _Float16 h[4]; uint2 u2; };

// gfx1250 async global->LDS copy (ASYNCcnt-tracked), GV addressing:
//   VDST = per-lane LDS byte address, VADDR = per-lane 64-bit global address.
#if defined(__HIP_DEVICE_COMPILE__)
#define ASYNC_LDS_B128(ldsaddr, gaddr)                                          \
    asm volatile("global_load_async_to_lds_b128 %0, %1, off"                    \
                 :: "v"(ldsaddr), "v"(gaddr) : "memory")
#define WAIT_ASYNC() asm volatile("s_wait_asynccnt 0x0" ::: "memory")
#else
#define ASYNC_LDS_B128(ldsaddr, gaddr) ((void)(ldsaddr), (void)(gaddr))
#define WAIT_ASYNC() ((void)0)
#endif

// D = scale * (A x B') + bias, accumulated in f32 via v_wmma_f32_16x16x32_f16.
// Double-buffered LDS pipeline: issue next tile (async b128 copies for f16
// sources, vectorized load+cvt for f32 sources) before computing current tile.
//   A: row-major [M x K], stride lda.
//   B': [K x N]. !B_NCONTIG: mem[n*ldb + k] (x@W.T / Q@K.T). B_NCONTIG:
//       mem[k*ldb + n] (attention@V, transposed during stage).
//   Output f32 (Cf) or f16 (Ch), stride ldc.  Per-z offs: (z/16)*So + (z%16)*Si.
template<int BM, int BN, int WAVES_M, int WAVES_N, int WTM, int WTN,
         typename TA, typename TB, bool B_NCONTIG, bool OUT_F16>
__global__ __launch_bounds__(256) void wmma_gemm(
    const TA* __restrict__ Ag, int lda, long long aSo, long long aSi,
    const TB* __restrict__ Bg, int ldb, long long bSo, long long bSi,
    float* __restrict__ Cf, _Float16* __restrict__ Ch, int ldc,
    long long cSo, long long cSi,
    const float* __restrict__ bias, float scale, int K)
{
    constexpr int  BK   = 32;
    constexpr int  STR  = BK + 16;        // 48 f16 = 96B rows: b128 stays 16B-aligned
    constexpr bool ASYNC_A = (sizeof(TA) == 2);
    constexpr bool ASYNC_B = (!B_NCONTIG) && (sizeof(TB) == 2);
    constexpr int  QPR4 = BK / 4;         // float4 chunks per row (8)
    constexpr int  QPR8 = BK / 8;         // b128 chunks per row (4)
    constexpr int  AIT  = (BM * QPR4) / 256;   // cvt-stage iters for A
    constexpr int  BIT  = (BN * QPR4) / 256;   // cvt-stage iters for B
    constexpr int  BTR  = (BN * BK) / 256;     // transpose-stage iters for B
    static_assert(BM == WAVES_M * WTM * 16, "tile mismatch M");
    static_assert(BN == WAVES_N * WTN * 16, "tile mismatch N");
    static_assert(WAVES_M * WAVES_N == 8, "8 waves of 32");

    __shared__ _Float16 sA[2][BM * STR];
    __shared__ _Float16 sB[2][BN * STR];

    const int z = blockIdx.z;
    const long long zo = z / N_HEADS_, zi = z % N_HEADS_;
    Ag += zo * aSo + zi * aSi;
    Bg += zo * bSo + zi * bSi;
    const long long coff = zo * cSo + zi * cSi;

    const int t     = threadIdx.x & 255;   // provably bounded -> unguarded unrolls
    const int lane  = t & 31;
    const int wid   = t >> 5;
    const int wm    = wid / WAVES_N;
    const int wn    = wid % WAVES_N;
    const int l16   = lane & 15;
    const int khalf = lane >> 4;

    const long long m0 = (long long)blockIdx.y * BM;
    const long long n0 = (long long)blockIdx.x * BN;

    v8f acc[WTM][WTN] = {};

    auto stageA_async = [&](int kk, int b) {
        #pragma unroll
        for (int e = t; e < BM * QPR8; e += 256) {
            const int m = e / QPR8, kq = e % QPR8;
            const uint32_t la = (uint32_t)(uintptr_t)&sA[b][m * STR + kq * 8];
            const uint64_t ga =
                (uint64_t)(uintptr_t)&Ag[(m0 + m) * (long long)lda + kk + kq * 8];
            ASYNC_LDS_B128(la, ga);
        }
    };
    auto stageB_async = [&](int kk, int b) {
        #pragma unroll
        for (int e = t; e < BN * QPR8; e += 256) {
            const int n = e / QPR8, kq = e % QPR8;
            const uint32_t la = (uint32_t)(uintptr_t)&sB[b][n * STR + kq * 8];
            const uint64_t ga =
                (uint64_t)(uintptr_t)&Bg[(n0 + n) * (long long)ldb + kk + kq * 8];
            ASYNC_LDS_B128(la, ga);
        }
    };
    auto stageA_cvt = [&](int kk, int b) {
        #pragma unroll
        for (int it = 0; it < AIT; ++it) {
            const int e = t + it * 256, m = e / QPR4, kq = e % QPR4;
            const float4 f = *(const float4*)&Ag[(m0 + m) * (long long)lda + kk + kq * 4];
            Cv4 cv;
            cv.h[0] = (_Float16)f.x; cv.h[1] = (_Float16)f.y;
            cv.h[2] = (_Float16)f.z; cv.h[3] = (_Float16)f.w;
            *(uint2*)&sA[b][m * STR + kq * 4] = cv.u2;
        }
    };
    auto stageB_cvt = [&](int kk, int b) {
        #pragma unroll
        for (int it = 0; it < BIT; ++it) {
            const int e = t + it * 256, n = e / QPR4, kq = e % QPR4;
            const float4 f = *(const float4*)&Bg[(n0 + n) * (long long)ldb + kk + kq * 4];
            Cv4 cv;
            cv.h[0] = (_Float16)f.x; cv.h[1] = (_Float16)f.y;
            cv.h[2] = (_Float16)f.z; cv.h[3] = (_Float16)f.w;
            *(uint2*)&sB[b][n * STR + kq * 4] = cv.u2;
        }
    };
    auto stageB_tr = [&](int kk, int b) {
        #pragma unroll
        for (int it = 0; it < BTR; ++it) {
            const int e = t + it * 256, kc = e / BN, n = e % BN;
            sB[b][n * STR + kc] = (_Float16)Bg[(long long)(kk + kc) * ldb + (n0 + n)];
        }
    };

    // ---- prologue: stage first tile into buffer 0 ----
    if (ASYNC_A) stageA_async(0, 0); else stageA_cvt(0, 0);
    if (B_NCONTIG) stageB_tr(0, 0);
    else if (ASYNC_B) stageB_async(0, 0);
    else stageB_cvt(0, 0);
    if (ASYNC_A || ASYNC_B) WAIT_ASYNC();
    __syncthreads();

    int buf = 0;
    for (int k0 = 0; k0 < K; k0 += BK) {
        const int  kn   = k0 + BK;
        const bool more = kn < K;

        float4 fa[AIT ? AIT : 1];
        float4 fb[BIT ? BIT : 1];
        TB     hb[BTR ? BTR : 1];

        // ---- issue next tile's global traffic before computing ----
        if (more) {
            if (ASYNC_A) stageA_async(kn, buf ^ 1);
            else {
                #pragma unroll
                for (int it = 0; it < AIT; ++it) {
                    const int e = t + it * 256, m = e / QPR4, kq = e % QPR4;
                    fa[it] = *(const float4*)&Ag[(m0 + m) * (long long)lda + kn + kq * 4];
                }
            }
            if (B_NCONTIG) {
                #pragma unroll
                for (int it = 0; it < BTR; ++it) {
                    const int e = t + it * 256, kc = e / BN, n = e % BN;
                    hb[it] = Bg[(long long)(kn + kc) * ldb + (n0 + n)];
                }
            } else if (ASYNC_B) stageB_async(kn, buf ^ 1);
            else {
                #pragma unroll
                for (int it = 0; it < BIT; ++it) {
                    const int e = t + it * 256, n = e / QPR4, kq = e % QPR4;
                    fb[it] = *(const float4*)&Bg[(n0 + n) * (long long)ldb + kn + kq * 4];
                }
            }
        }

        // ---- compute current tile: fragments per ISA wave32 layouts ----
        Frag16 af[WTM], bf[WTN];
        #pragma unroll
        for (int r = 0; r < WTM; ++r) {
            // A 16x32 f16: lanes 0-15 K {0..7,16..23}; lanes 16-31 K {8..15,24..31}
            const _Float16* row = &sA[buf][((wm * WTM + r) * 16 + l16) * STR + khalf * 8];
            #pragma unroll
            for (int i = 0; i < 4; ++i) {
                af[r].u[i]     = *(const uint32_t*)&row[2 * i];
                af[r].u[4 + i] = *(const uint32_t*)&row[16 + 2 * i];
            }
        }
        #pragma unroll
        for (int c = 0; c < WTN; ++c) {
            // B 32x16 f16: lane = column n; lanes 0-15 K {0..7,16..23}; lanes 16-31 K {8..15,24..31}
            const _Float16* col = &sB[buf][((wn * WTN + c) * 16 + l16) * STR + khalf * 8];
            #pragma unroll
            for (int i = 0; i < 4; ++i) {
                bf[c].u[i]     = *(const uint32_t*)&col[2 * i];
                bf[c].u[4 + i] = *(const uint32_t*)&col[16 + 2 * i];
            }
        }
        #pragma unroll
        for (int r = 0; r < WTM; ++r)
            #pragma unroll
            for (int c = 0; c < WTN; ++c)
                acc[r][c] = __builtin_amdgcn_wmma_f32_16x16x32_f16(
                    false, af[r].v, false, bf[c].v, (short)0, acc[r][c], false, false);

        // ---- finish staging next tile into the other buffer ----
        if (more) {
            if (!ASYNC_A) {
                #pragma unroll
                for (int it = 0; it < AIT; ++it) {
                    const int e = t + it * 256, m = e / QPR4, kq = e % QPR4;
                    Cv4 cv;
                    cv.h[0] = (_Float16)fa[it].x; cv.h[1] = (_Float16)fa[it].y;
                    cv.h[2] = (_Float16)fa[it].z; cv.h[3] = (_Float16)fa[it].w;
                    *(uint2*)&sA[buf ^ 1][m * STR + kq * 4] = cv.u2;
                }
            }
            if (B_NCONTIG) {
                #pragma unroll
                for (int it = 0; it < BTR; ++it) {
                    const int e = t + it * 256, kc = e / BN, n = e % BN;
                    sB[buf ^ 1][n * STR + kc] = (_Float16)hb[it];
                }
            } else if (!ASYNC_B) {
                #pragma unroll
                for (int it = 0; it < BIT; ++it) {
                    const int e = t + it * 256, n = e / QPR4, kq = e % QPR4;
                    Cv4 cv;
                    cv.h[0] = (_Float16)fb[it].x; cv.h[1] = (_Float16)fb[it].y;
                    cv.h[2] = (_Float16)fb[it].z; cv.h[3] = (_Float16)fb[it].w;
                    *(uint2*)&sB[buf ^ 1][n * STR + kq * 4] = cv.u2;
                }
            }
            if (ASYNC_A || ASYNC_B) WAIT_ASYNC();
        }
        __syncthreads();
        buf ^= 1;
    }

    // ---- writeback: VGPR i -> row base + khalf*8 + i, col = lane&15 ----
    #pragma unroll
    for (int r = 0; r < WTM; ++r) {
        const long long mrow = m0 + (wm * WTM + r) * 16 + khalf * 8;
        #pragma unroll
        for (int c = 0; c < WTN; ++c) {
            const long long ncol = n0 + (wn * WTN + c) * 16 + l16;
            const float bv = bias ? bias[ncol] : 0.0f;
            #pragma unroll
            for (int i = 0; i < 8; ++i) {
                const float val = acc[r][c][i] * scale + bv;
                if (OUT_F16) Ch[coff + (mrow + i) * (long long)ldc + ncol] = (_Float16)val;
                else         Cf[coff + (mrow + i) * (long long)ldc + ncol] = val;
            }
        }
    }
}

// In-place row softmax over S=2048 columns; one 256-thread block per row.
__global__ __launch_bounds__(256) void softmax_rows(float* __restrict__ attn)
{
    constexpr int PER = SEQ_ / 256;   // 8
    float* p = attn + (long long)blockIdx.x * SEQ_;
    __shared__ float red[256];
    const int t = threadIdx.x & 255;

    float v[PER];
    float m = -3.402823466e38f;
    #pragma unroll
    for (int i = 0; i < PER; ++i) { v[i] = p[t + i * 256]; m = fmaxf(m, v[i]); }
    red[t] = m; __syncthreads();
    for (int s = 128; s > 0; s >>= 1) { if (t < s) red[t] = fmaxf(red[t], red[t + s]); __syncthreads(); }
    m = red[0]; __syncthreads();

    float sum = 0.0f;
    #pragma unroll
    for (int i = 0; i < PER; ++i) { v[i] = __expf(v[i] - m); sum += v[i]; }
    red[t] = sum; __syncthreads();
    for (int s = 128; s > 0; s >>= 1) { if (t < s) red[t] += red[t + s]; __syncthreads(); }
    const float inv = 1.0f / red[0];
    #pragma unroll
    for (int i = 0; i < PER; ++i) p[t + i * 256] = v[i] * inv;
}

extern "C" void kernel_launch(void* const* d_in, const int* in_sizes, int n_in,
                              void* d_out, int out_size, void* d_ws, size_t ws_size,
                              hipStream_t stream)
{
    (void)in_sizes; (void)n_in; (void)out_size; (void)ws_size;

    const float* q  = (const float*)d_in[0];
    const float* k  = (const float*)d_in[1];
    const float* v  = (const float*)d_in[2];
    const float* Wq = (const float*)d_in[3];
    const float* bq = (const float*)d_in[4];
    const float* Wk = (const float*)d_in[5];
    const float* bk = (const float*)d_in[6];
    const float* Wv = (const float*)d_in[7];
    const float* bv = (const float*)d_in[8];
    const float* Wo = (const float*)d_in[9];
    const float* bo = (const float*)d_in[10];

    float* out  = (float*)d_out;
    float* attn = out + (size_t)BATCH_ * SEQ_ * D_MODEL_;   // [B,H,S,S]

    const size_t NE = (size_t)BATCH_ * SEQ_ * D_MODEL_;     // 8.39M elems
    _Float16* Qh = (_Float16*)d_ws;      // projected Q, f16
    _Float16* Kh = Qh + NE;              // projected K, f16
    _Float16* Vh = Kh + NE;              // projected V, f16
    _Float16* Ch = Vh + NE;              // attention context, f16 (~67MB total ws)

    const dim3 blk(256, 1, 1);
    const int M = BATCH_ * SEQ_;         // 8192 rows

    // 1) Q/K/V projections: [8192x1024] = x @ W^T + b, f16 outputs
    {
        dim3 grid(D_MODEL_ / 128, M / 128, 1);
        wmma_gemm<128,128,4,2,2,4, float, float, false, true><<<grid, blk, 0, stream>>>(
            q, D_MODEL_, 0, 0, Wq, D_MODEL_, 0, 0, nullptr, Qh, D_MODEL_, 0, 0, bq, 1.0f, D_MODEL_);
        wmma_gemm<128,128,4,2,2,4, float, float, false, true><<<grid, blk, 0, stream>>>(
            k, D_MODEL_, 0, 0, Wk, D_MODEL_, 0, 0, nullptr, Kh, D_MODEL_, 0, 0, bk, 1.0f, D_MODEL_);
        wmma_gemm<128,128,4,2,2,4, float, float, false, true><<<grid, blk, 0, stream>>>(
            v, D_MODEL_, 0, 0, Wv, D_MODEL_, 0, 0, nullptr, Vh, D_MODEL_, 0, 0, bv, 1.0f, D_MODEL_);
    }
    // 2) scores = Q K^T / sqrt(d_k), fp32 into attention region (async f16 staging)
    {
        dim3 grid(SEQ_ / 128, SEQ_ / 128, BATCH_ * N_HEADS_);
        const long long qSo = (long long)SEQ_ * D_MODEL_, qSi = D_K_;
        const long long cSo = (long long)N_HEADS_ * SEQ_ * SEQ_, cSi = (long long)SEQ_ * SEQ_;
        wmma_gemm<128,128,4,2,2,4, _Float16, _Float16, false, false><<<grid, blk, 0, stream>>>(
            Qh, D_MODEL_, qSo, qSi, Kh, D_MODEL_, qSo, qSi,
            attn, nullptr, SEQ_, cSo, cSi, nullptr, 0.125f, D_K_);
    }
    // 3) softmax rows, in place
    softmax_rows<<<dim3(BATCH_ * N_HEADS_ * SEQ_, 1, 1), blk, 0, stream>>>(attn);
    // 4) context = attention @ V
    {
        dim3 grid(D_K_ / 64, SEQ_ / 128, BATCH_ * N_HEADS_);
        const long long aSo = (long long)N_HEADS_ * SEQ_ * SEQ_, aSi = (long long)SEQ_ * SEQ_;
        const long long vSo = (long long)SEQ_ * D_MODEL_, vSi = D_K_;
        wmma_gemm<128,64,4,2,2,2, float, _Float16, true, true><<<grid, blk, 0, stream>>>(
            attn, SEQ_, aSo, aSi, Vh, D_MODEL_, vSo, vSi,
            nullptr, Ch, D_MODEL_, vSo, vSi, nullptr, 1.0f, SEQ_);
    }
    // 5) out = context @ Wo^T + bo (async f16 A staging)
    {
        dim3 grid(D_MODEL_ / 128, M / 128, 1);
        wmma_gemm<128,128,4,2,2,4, _Float16, float, false, false><<<grid, blk, 0, stream>>>(
            Ch, D_MODEL_, 0, 0, Wo, D_MODEL_, 0, 0, out, nullptr, D_MODEL_, 0, 0, bo, 1.0f, D_MODEL_);
    }
}
